// PerformerAttention_3693671875324
// MI455X (gfx1250) — matrix operand order repaired and checked by reading, not compiled, not run on hardware
//
#include <hip/hip_runtime.h>

#define B_    4
#define N_    4096
#define DIM_  1024
#define HEADS_ 16
#define DHEAD_ 64
#define NRF_  64
#define STAB_ 1e-6f

typedef _Float16 h4  __attribute__((ext_vector_type(4)));
typedef _Float16 h8  __attribute__((ext_vector_type(8)));
typedef _Float16 v16h __attribute__((ext_vector_type(16)));
typedef float    vf4 __attribute__((ext_vector_type(4)));
typedef float    v8f __attribute__((ext_vector_type(8)));

__device__ __forceinline__ v8f zero_v8f() {
  v8f z;
#pragma unroll
  for (int i = 0; i < 8; ++i) z[i] = 0.f;
  return z;
}

__device__ __forceinline__ v8f wmma_f16(v16h a, v16h b, v8f c) {
  // D = A(16x32 f16) * B(32x16 f16) + C(16x16 f32)
  return __builtin_amdgcn_wmma_f32_16x16x32_f16(false, a, false, b, (short)0, c,
                                                false, false);
}

// A fragment: tile is row-major [16][K>=32] f16 in LDS, ldh multiple of 8 halves.
// lane&15 = M row; lanes 0-15 hold K 0..7 & 16..23, lanes 16-31 hold K 8..15 & 24..31.
__device__ __forceinline__ v16h load_A_frag(const _Float16* tile, int ldh, int lane) {
  const int m = lane & 15;
  const int koff = (lane >> 4) << 3;  // 0 or 8
  const h8* p = (const h8*)(tile + m * ldh + koff);
  h8 lo = p[0];   // K = koff .. koff+7
  h8 hi = p[2];   // K = koff+16 .. koff+23
  v16h r;
#pragma unroll
  for (int j = 0; j < 8; ++j) { r[j] = lo[j]; r[8 + j] = hi[j]; }
  return r;
}

// B fragment: tileT stored K-contiguous: tileT[n][k] (B transposed), ldh mult of 8.
// lane&15 = N col; lanes 0-15 hold K 0..7 & 16..23, lanes 16-31 hold K 8..15 & 24..31.
__device__ __forceinline__ v16h load_B_frag(const _Float16* tileT, int ldh, int lane) {
  const int n = lane & 15;
  const int koff = (lane >> 4) << 3;  // 0 or 8
  const h8* p = (const h8*)(tileT + n * ldh + koff);
  h8 lo = p[0];
  h8 hi = p[2];
  v16h r;
#pragma unroll
  for (int j = 0; j < 8; ++j) { r[j] = lo[j]; r[8 + j] = hi[j]; }
  return r;
}

__device__ __forceinline__ void ld4h(const float* p, _Float16* d) {
  vf4 v = *(const vf4*)p;
#pragma unroll
  for (int j = 0; j < 4; ++j) d[j] = (_Float16)v[j];
}
__device__ __forceinline__ void ld4h(const _Float16* p, _Float16* d) {
  h4 v = *(const h4*)p;
#pragma unroll
  for (int j = 0; j < 4; ++j) d[j] = v[j];
}

// ---------------------------------------------------------------------------
// Generic tiled WMMA GEMM: C[M,N] = A[M,K] * B[K,N] (+bias), fp32 accumulate.
// Block tile 256x128, K-step 32, 256 threads = 8 waves (4x2), wave tile 64x64:
// 16 accumulators, 16 ds_load_b128 per 16 WMMAs per K-step (ds:wmma = 1.0).
// A is f16 or f32 in global (converted on stage); B is f32 in global.
// ---------------------------------------------------------------------------
template <typename AT, bool OUT_F16, bool BIAS>
__global__ void __launch_bounds__(256)
gemm_wmma(const AT* __restrict__ A, int lda,
          const float* __restrict__ Bm, int ldb,
          void* __restrict__ Cp, int ldc,
          const float* __restrict__ bias, int K) {
  __shared__ __align__(16) _Float16 As[256][48];  // row-major, K contiguous
  __shared__ __align__(16) _Float16 Bt[128][48];  // [n][k] (K-transposed)

  const int t = threadIdx.x;
  const int lane = t & 31;
  const int wid = t >> 5;
  const int wm = (wid >> 1) * 64;   // wave M offset in block (4 waves down M)
  const int wn = (wid & 1) * 64;    // wave N offset in block (2 waves across N)
  const int m0 = blockIdx.y * 256;
  const int n0 = blockIdx.x * 128;

  const int ar = t >> 3;            // A stage: row 0..31 (+32*i)
  const int ac = (t & 7) * 4;       // A stage: col group
  const int br = t >> 5;            // B stage: row 0..7 (+8*i)
  const int bc = (t & 31) * 4;      // B stage: col group

  v8f acc[4][4];
#pragma unroll
  for (int i = 0; i < 4; ++i)
#pragma unroll
    for (int j = 0; j < 4; ++j) acc[i][j] = zero_v8f();

  for (int kt = 0; kt < K; kt += 32) {
    // Stage to registers first (overlaps with previous tile's compute).
    __align__(8) _Float16 areg[8][4];
#pragma unroll
    for (int i = 0; i < 8; ++i)
      ld4h(A + (size_t)(m0 + ar + i * 32) * lda + kt + ac, areg[i]);
    __align__(8) _Float16 breg[4][4];
#pragma unroll
    for (int i = 0; i < 4; ++i)
      ld4h(Bm + (size_t)(kt + br + i * 8) * ldb + n0 + bc, breg[i]);

    __syncthreads();  // previous compute done reading LDS
#pragma unroll
    for (int i = 0; i < 8; ++i)
      *(h4*)&As[ar + i * 32][ac] = *(const h4*)areg[i];
#pragma unroll
    for (int i = 0; i < 4; ++i)
#pragma unroll
      for (int j = 0; j < 4; ++j)
        Bt[bc + j][br + i * 8] = breg[i][j];
    __syncthreads();

    if (kt + 32 < K) {  // uniform; speculative prefetch of next K tiles
      __builtin_prefetch(A + (size_t)(m0 + ar) * lda + kt + 32 + ac, 0, 1);
      __builtin_prefetch(Bm + (size_t)(kt + 32 + br) * ldb + n0 + bc, 0, 1);
    }

    v16h af[4];
#pragma unroll
    for (int mt = 0; mt < 4; ++mt)
      af[mt] = load_A_frag(&As[wm + mt * 16][0], 48, lane);
#pragma unroll
    for (int nt = 0; nt < 4; ++nt) {
      v16h bf = load_B_frag(&Bt[wn + nt * 16][0], 48, lane);
#pragma unroll
      for (int mt = 0; mt < 4; ++mt)
        acc[mt][nt] = wmma_f16(af[mt], bf, acc[mt][nt]);
    }
  }

  const int rbase = (lane >> 4) << 3;
  const int cl = lane & 15;
#pragma unroll
  for (int mt = 0; mt < 4; ++mt) {
#pragma unroll
    for (int nt = 0; nt < 4; ++nt) {
      const int col = n0 + wn + nt * 16 + cl;
      const float bv = BIAS ? bias[col] : 0.f;
#pragma unroll
      for (int j = 0; j < 8; ++j) {
        const int row = m0 + wm + mt * 16 + rbase + j;
        const float val = acc[mt][nt][j] + bv;
        if (OUT_F16)
          ((_Float16*)Cp)[(size_t)row * ldc + col] = (_Float16)val;
        else
          ((float*)Cp)[(size_t)row * ldc + col] = val;
      }
    }
  }
}

// ---------------------------------------------------------------------------
// phi = relu( q_or_k[b,h] @ RF[h] ) : per-(b,h,z) 4096x64x64, fused ReLU.
// grid (32, 64, 2); 256 threads = 8 waves; wave = 16 rows x 64 cols.
// ---------------------------------------------------------------------------
__global__ void __launch_bounds__(256)
phi_kernel(const _Float16* __restrict__ qkv,  // [B*N][3072]
           const float* __restrict__ RF,      // [H][64][64]
           _Float16* __restrict__ phi_q,
           _Float16* __restrict__ phi_k) {
  __shared__ __align__(16) _Float16 As[128][80];
  __shared__ __align__(16) _Float16 Bt[64][80];

  const int t = threadIdx.x;
  const int lane = t & 31;
  const int wid = t >> 5;
  const int n0 = blockIdx.x * 128;
  const int bh = blockIdx.y;
  const int h = bh & 15;
  const int isK = blockIdx.z;
  const int colbase = isK * (HEADS_ * DHEAD_) + h * DHEAD_;

  {  // stage A: 128 rows x 64 halves of q or k
    const int row = t >> 1;
    const int c0 = (t & 1) * 32;
    const _Float16* src =
        qkv + (size_t)(bh >> 4) * N_ * 3072 + (size_t)(n0 + row) * 3072 + colbase + c0;
#pragma unroll
    for (int i = 0; i < 4; ++i)
      *(h8*)&As[row][c0 + i * 8] = *(const h8*)(src + i * 8);
  }
  {  // stage B: RF[h] (64x64 f32), transposed to Bt[r][d], cvt to f16
    const int d = t >> 2;
    const int r0 = (t & 3) * 16;
    const float* src = RF + ((size_t)h * 64 + d) * 64 + r0;
#pragma unroll
    for (int i = 0; i < 4; ++i) {
      vf4 v = *(const vf4*)(src + i * 4);
#pragma unroll
      for (int j = 0; j < 4; ++j) Bt[r0 + i * 4 + j][d] = (_Float16)v[j];
    }
  }
  __syncthreads();

  v8f acc[4];
#pragma unroll
  for (int i = 0; i < 4; ++i) acc[i] = zero_v8f();

#pragma unroll
  for (int kt = 0; kt < 64; kt += 32) {
    v16h a = load_A_frag(&As[wid * 16][kt], 80, lane);
#pragma unroll
    for (int nt = 0; nt < 4; ++nt) {
      v16h b = load_B_frag(&Bt[nt * 16][kt], 80, lane);
      acc[nt] = wmma_f16(a, b, acc[nt]);
    }
  }

  _Float16* dst = (isK ? phi_k : phi_q) + (size_t)bh * N_ * NRF_;
  const int rbase = (lane >> 4) << 3;
  const int cl = lane & 15;
#pragma unroll
  for (int nt = 0; nt < 4; ++nt) {
    const int col = nt * 16 + cl;
#pragma unroll
    for (int j = 0; j < 8; ++j) {
      const int row = n0 + wid * 16 + rbase + j;
      float v = acc[nt][j];
      v = v > 0.f ? v : 0.f;  // ReLU
      dst[(size_t)row * NRF_ + col] = (_Float16)v;
    }
  }
}

// k_sum[bh][r] = sum_n phi_k[bh][n][r]  (fp32)
__global__ void __launch_bounds__(256)
ksum_kernel(const _Float16* __restrict__ phi_k, float* __restrict__ ksum) {
  __shared__ float red[256];
  const int bh = blockIdx.x;
  const int t = threadIdx.x;
  const int r = t & 63;
  const int part = t >> 6;
  const _Float16* src = phi_k + (size_t)bh * N_ * NRF_;
  float s = 0.f;
  const int n_beg = part * (N_ / 4), n_end = n_beg + (N_ / 4);
  for (int n = n_beg; n < n_end; ++n) s += (float)src[(size_t)n * NRF_ + r];
  red[t] = s;
  __syncthreads();
  if (t < 64)
    ksum[(size_t)bh * 64 + t] = red[t] + red[t + 64] + red[t + 128] + red[t + 192];
}

// ---------------------------------------------------------------------------
// kv[bh] = phi_k[bh]^T @ v[bh] : 64x64 over K=4096. 128 threads = 4 waves,
// wave = 16 r-rows x 64 d-cols; both operands transposed into LDS.
// ---------------------------------------------------------------------------
__global__ void __launch_bounds__(128)
kv_kernel(const _Float16* __restrict__ phi_k,
          const _Float16* __restrict__ qkv,
          _Float16* __restrict__ kv) {
  __shared__ __align__(16) _Float16 At[64][48];  // [r][n]
  __shared__ __align__(16) _Float16 Bt[64][48];  // [d][n]
  const int t = threadIdx.x;
  const int lane = t & 31;
  const int wid = t >> 5;  // 0..3
  const int bh = blockIdx.x;
  const int b = bh >> 4, h = bh & 15;
  const _Float16* pk = phi_k + (size_t)bh * N_ * NRF_;
  const _Float16* vsrc = qkv + (size_t)b * N_ * 3072 + 2 * HEADS_ * DHEAD_ + h * DHEAD_;

  const int sn = t >> 2;         // n in chunk 0..31
  const int sc0 = (t & 3) * 16;  // column base 0..48

  v8f acc[4];
#pragma unroll
  for (int i = 0; i < 4; ++i) acc[i] = zero_v8f();

  for (int n0 = 0; n0 < N_; n0 += 32) {
    h8 a0 = *(const h8*)(pk + (size_t)(n0 + sn) * NRF_ + sc0);
    h8 a1 = *(const h8*)(pk + (size_t)(n0 + sn) * NRF_ + sc0 + 8);
    h8 b0 = *(const h8*)(vsrc + (size_t)(n0 + sn) * 3072 + sc0);
    h8 b1 = *(const h8*)(vsrc + (size_t)(n0 + sn) * 3072 + sc0 + 8);
    __syncthreads();
#pragma unroll
    for (int j = 0; j < 8; ++j) {
      At[sc0 + j][sn] = a0[j];
      At[sc0 + 8 + j][sn] = a1[j];
      Bt[sc0 + j][sn] = b0[j];
      Bt[sc0 + 8 + j][sn] = b1[j];
    }
    __syncthreads();
    v16h a = load_A_frag(&At[wid * 16][0], 48, lane);
#pragma unroll
    for (int nt = 0; nt < 4; ++nt) {
      v16h bf = load_B_frag(&Bt[nt * 16][0], 48, lane);
      acc[nt] = wmma_f16(a, bf, acc[nt]);
    }
  }

  const int rbase = (lane >> 4) << 3;
  const int cl = lane & 15;
  _Float16* dst = kv + (size_t)bh * 64 * 64;
#pragma unroll
  for (int nt = 0; nt < 4; ++nt) {
    const int col = nt * 16 + cl;
#pragma unroll
    for (int j = 0; j < 8; ++j) {
      const int row = wid * 16 + rbase + j;
      dst[row * 64 + col] = (_Float16)acc[nt][j];
    }
  }
}

// ---------------------------------------------------------------------------
// out[bh] = (phi_q[bh] @ kv[bh]) / (phi_q[bh].k_sum[bh] + eps), written as f16
// directly into att[b, n, h*64 + d] layout for the final projection GEMM.
// ---------------------------------------------------------------------------
__global__ void __launch_bounds__(256)
outnorm_kernel(const _Float16* __restrict__ phi_q,
               const _Float16* __restrict__ kv,
               const float* __restrict__ ksum,
               _Float16* __restrict__ att) {
  __shared__ __align__(16) _Float16 As[128][80];
  __shared__ __align__(16) _Float16 Bt[64][80];
  __shared__ float ksum_s[64];
  __shared__ float norm_s[128];
  const int t = threadIdx.x;
  const int lane = t & 31;
  const int wid = t >> 5;
  const int n0 = blockIdx.x * 128;
  const int bh = blockIdx.y;
  const int b = bh >> 4, h = bh & 15;
  const _Float16* pq = phi_q + (size_t)bh * N_ * NRF_;

  {  // stage A: phi_q rows
    const int row = t >> 1;
    const int c0 = (t & 1) * 32;
    const _Float16* src = pq + (size_t)(n0 + row) * NRF_ + c0;
#pragma unroll
    for (int i = 0; i < 4; ++i)
      *(h8*)&As[row][c0 + i * 8] = *(const h8*)(src + i * 8);
  }
  {  // stage B: kv[r][d] -> Bt[d][r]
    const int r = t >> 2;
    const int d0 = (t & 3) * 16;
    const _Float16* src = kv + (size_t)bh * 4096 + r * 64 + d0;
    h8 v0 = *(const h8*)(src);
    h8 v1 = *(const h8*)(src + 8);
#pragma unroll
    for (int j = 0; j < 8; ++j) {
      Bt[d0 + j][r] = v0[j];
      Bt[d0 + 8 + j][r] = v1[j];
    }
  }
  if (t < 64) ksum_s[t] = ksum[(size_t)bh * 64 + t];
  __syncthreads();

  if (t < 128) {  // normalizer per output row (reads staged A tile)
    float s = 0.f;
#pragma unroll
    for (int r = 0; r < 64; ++r) s += (float)As[t][r] * ksum_s[r];
    norm_s[t] = s + STAB_;
  }

  v8f acc[4];
#pragma unroll
  for (int i = 0; i < 4; ++i) acc[i] = zero_v8f();

#pragma unroll
  for (int kt = 0; kt < 64; kt += 32) {
    v16h a = load_A_frag(&As[wid * 16][kt], 80, lane);
#pragma unroll
    for (int nt = 0; nt < 4; ++nt) {
      v16h bf = load_B_frag(&Bt[nt * 16][kt], 80, lane);
      acc[nt] = wmma_f16(a, bf, acc[nt]);
    }
  }
  __syncthreads();  // norm_s visible to all

  const int rbase = (lane >> 4) << 3;
  const int cl = lane & 15;
  _Float16* dst = att + (size_t)b * N_ * DIM_ + (size_t)h * DHEAD_;
#pragma unroll
  for (int nt = 0; nt < 4; ++nt) {
    const int col = nt * 16 + cl;
#pragma unroll
    for (int j = 0; j < 8; ++j) {
      const int lrow = wid * 16 + rbase + j;
      const float v = acc[nt][j] / norm_s[lrow];
      dst[(size_t)(n0 + lrow) * DIM_ + col] = (_Float16)v;
    }
  }
}

extern "C" void kernel_launch(void* const* d_in, const int* in_sizes, int n_in,
                              void* d_out, int out_size, void* d_ws, size_t ws_size,
                              hipStream_t stream) {
  const float* x     = (const float*)d_in[0];  // [4,4096,1024]
  const float* W_qkv = (const float*)d_in[1];  // [1024,3072]
  const float* RF    = (const float*)d_in[2];  // [16,64,64]
  const float* W_out = (const float*)d_in[3];  // [1024,1024]
  const float* b_out = (const float*)d_in[4];  // [1024]

  // Workspace layout (all f16 unless noted):
  char* ws = (char*)d_ws;
  const size_t qkv_bytes = (size_t)16384 * 3072 * 2;    // 96 MB
  const size_t phi_bytes = (size_t)64 * 4096 * 64 * 2;  // 32 MB each
  const size_t att_bytes = (size_t)16384 * 1024 * 2;    // 32 MB
  _Float16* qkv_h = (_Float16*)ws;
  _Float16* phi_q = (_Float16*)(ws + qkv_bytes);
  _Float16* phi_k = (_Float16*)(ws + qkv_bytes + phi_bytes);
  _Float16* att   = (_Float16*)(ws + qkv_bytes + 2 * phi_bytes);
  _Float16* kvb   = (_Float16*)(ws + qkv_bytes + 2 * phi_bytes + att_bytes);
  float*    ksum  = (float*)(ws + qkv_bytes + 2 * phi_bytes + att_bytes +
                             (size_t)64 * 64 * 64 * 2);

  // 1) qkv = x @ W_qkv  (M=16384, N=3072, K=1024), f16 out
  gemm_wmma<float, true, false>
      <<<dim3(3072 / 128, 16384 / 256), 256, 0, stream>>>(
          x, DIM_, W_qkv, 3072, qkv_h, 3072, nullptr, DIM_);

  // 2) phi_q / phi_k = relu(q|k @ RF[h])
  phi_kernel<<<dim3(N_ / 128, B_ * HEADS_, 2), 256, 0, stream>>>(
      qkv_h, RF, phi_q, phi_k);

  // 3) k_sum
  ksum_kernel<<<dim3(B_ * HEADS_), 256, 0, stream>>>(phi_k, ksum);

  // 4) kv = phi_k^T @ v
  kv_kernel<<<dim3(B_ * HEADS_), 128, 0, stream>>>(phi_k, qkv_h, kvb);

  // 5) att = (phi_q @ kv) / (phi_q . k_sum + eps)
  outnorm_kernel<<<dim3(N_ / 128, B_ * HEADS_), 256, 0, stream>>>(
      phi_q, kvb, ksum, att);

  // 6) y = att @ W_out + b_out  (fp32 out)
  gemm_wmma<_Float16, false, true>
      <<<dim3(1024 / 128, 16384 / 256), 256, 0, stream>>>(
          att, DIM_, W_out, DIM_, (float*)d_out, DIM_, b_out, DIM_);
}
